// MorphemeSegmenter_46420006535351
// MI455X (gfx1250) — matrix operand order repaired and checked by reading, not compiled, not run on hardware
//
#include <hip/hip_runtime.h>
#include <hip/hip_bf16.h>

typedef __attribute__((ext_vector_type(4)))  float    v4f;
typedef __attribute__((ext_vector_type(8)))  float    v8f;
typedef __attribute__((ext_vector_type(4)))  _Float16 v4h;
typedef __attribute__((ext_vector_type(8)))  _Float16 v8h;
typedef __attribute__((ext_vector_type(16))) _Float16 v16h;

union F16x16 { v16h v; v8h h[2]; };

#define CB   64      // batch
#define CS   512     // sequence
#define CE   512     // embed
#define CH   4       // heads
#define CDH  128     // head dim
#define CE2  256     // hidden for heads
#define CK   5       // max segments
#define CNEG (-1000000000.0f)
#define CEPS (1e-10f)

// ---------------------------------------------------------------------------
// f16-WMMA GEMM:  C[M,N] = A[M,K] * B[K,N] + bias (+residual)
// Template-specialized on A dtype / C dtype / residual so each instantiation
// is branch-free. LDS tiles padded to stride 40 halves (80B -> 20-bank
// stride, conflict-free fragment loads). A fragments: two 16B LDS loads.
// B tile staged column-major; staging writes are two 16B ds_store_b128 per
// thread. Block = 128 threads (4 waves), tile 64x64x32.
// ---------------------------------------------------------------------------
template <int A_HALF, int C_HALF, int HAS_RES>
__global__ __launch_bounds__(128) void gemm_wmma_kernel(
    const void* __restrict__ Aptr, const float* __restrict__ Bw,
    const float* __restrict__ bias, const float* __restrict__ residual,
    void* __restrict__ Cptr, int M, int N, int Kd)
{
    constexpr int LDT = 40;                            // padded tile stride
    __shared__ __align__(128) _Float16 As[64 * LDT];   // row-major [m][k]
    __shared__ __align__(128) _Float16 Bs[64 * LDT];   // col-major [n][k]
    const int tid  = threadIdx.x;
    const int lane = tid & 31;
    const int wave = tid >> 5;
    const int lo   = lane & 15;
    const int hi16 = lane & 16;       // 0 or 16
    const int hi8  = hi16 >> 1;       // 0 or 8
    const int m0 = blockIdx.y * 64;
    const int n0 = blockIdx.x * 64;

    v8f acc[4];
    for (int nt = 0; nt < 4; ++nt)
        for (int r = 0; r < 8; ++r) acc[nt][r] = 0.0f;

    // staging coordinates (fixed per thread)
    const int am = tid >> 1;             // A row 0..63
    const int aq = (tid & 1) * 16;       // A k-offset 0/16
    const int bn = tid & 63;             // B column 0..63
    const int bk = (tid >> 6) * 16;      // B k-offset 0/16

    for (int kt = 0; kt < Kd; kt += 32) {
        // ---- stage A tile (64x32) as f16: 16 halves per thread ----
        {
            size_t gi = (size_t)(m0 + am) * Kd + kt + aq;
            v8h h0, h1;
            if (A_HALF) {
                const _Float16* Ah = (const _Float16*)Aptr;
                h0 = *(const v8h*)(Ah + gi);
                h1 = *(const v8h*)(Ah + gi + 8);
                if (kt + 32 < Kd) __builtin_prefetch(Ah + gi + 32, 0, 1);
            } else {
                const float* Af = (const float*)Aptr;
                v4f f0 = *(const v4f*)(Af + gi);
                v4f f1 = *(const v4f*)(Af + gi + 4);
                v4f f2 = *(const v4f*)(Af + gi + 8);
                v4f f3 = *(const v4f*)(Af + gi + 12);
#pragma unroll
                for (int j = 0; j < 4; ++j) {
                    h0[j]     = (_Float16)f0[j];
                    h0[4 + j] = (_Float16)f1[j];
                    h1[j]     = (_Float16)f2[j];
                    h1[4 + j] = (_Float16)f3[j];
                }
                if (kt + 32 < Kd) __builtin_prefetch(Af + gi + 32, 0, 1);
            }
            *(v8h*)(As + am * LDT + aq)     = h0;
            *(v8h*)(As + am * LDT + aq + 8) = h1;
        }
        // ---- stage B tile (32x64) f32->f16, column-major ----
        {
            const float* bp = Bw + (size_t)(kt + bk) * N + n0 + bn;
            v8h b0, b1;
#pragma unroll
            for (int j = 0; j < 8; ++j) b0[j] = (_Float16)bp[(size_t)j * N];
#pragma unroll
            for (int j = 0; j < 8; ++j) b1[j] = (_Float16)bp[(size_t)(j + 8) * N];
            *(v8h*)(Bs + bn * LDT + bk)     = b0;
            *(v8h*)(Bs + bn * LDT + bk + 8) = b1;
            if (kt + 32 < Kd)
                __builtin_prefetch(bp + (size_t)32 * N, 0, 1);
        }
        __syncthreads();

        // ---- A fragment: two contiguous 16B LDS loads ----
        F16x16 af;
        const _Float16* arow = As + (wave * 16 + lo) * LDT;
        af.h[0] = *(const v8h*)(arow + hi8);
        af.h[1] = *(const v8h*)(arow + 16 + hi8);

#pragma unroll
        for (int nt = 0; nt < 4; ++nt) {
            // ---- B fragment: two contiguous 16B LDS loads ----
            F16x16 bf;
            const _Float16* brow = Bs + (nt * 16 + lo) * LDT;
            bf.h[0] = *(const v8h*)(brow + hi8);
            bf.h[1] = *(const v8h*)(brow + 16 + hi8);
            acc[nt] = __builtin_amdgcn_wmma_f32_16x16x32_f16(
                false, af.v, false, bf.v, (short)0, acc[nt], false, false);
        }
        __syncthreads();
    }

    // epilogue: C/D layout M = r + 8*(lane>=16), N = lane%16
    for (int nt = 0; nt < 4; ++nt) {
        for (int r = 0; r < 8; ++r) {
            int m = m0 + wave * 16 + r + hi8;
            int n = n0 + nt * 16 + lo;
            size_t ci = (size_t)m * N + n;
            float v = acc[nt][r] + bias[n];
            if (HAS_RES) v += residual[ci];
            if (C_HALF) ((_Float16*)Cptr)[ci] = (_Float16)v;
            else        ((float*)Cptr)[ci] = v;
        }
    }
}

// ---------------------------------------------------------------------------
// Transpose V head-slices: vT[b][h][d][s] = qkv[b][s][2E + h*128 + d]
// ---------------------------------------------------------------------------
__global__ __launch_bounds__(256) void vtrans_kernel(
    const _Float16* __restrict__ qkv, _Float16* __restrict__ vT)
{
    __shared__ _Float16 tile[64][72];
    const int bh = blockIdx.z;            // b*4 + h
    const int b  = bh >> 2, h = bh & 3;
    const int s0 = blockIdx.x * 64;
    const int d0 = blockIdx.y * 64;
    const int tid = threadIdx.x;
    for (int i = tid; i < 64 * 64; i += 256) {
        int sR = i >> 6, dR = i & 63;
        tile[sR][dR] =
            qkv[((size_t)b * CS + s0 + sR) * (3 * CE) + 2 * CE + h * CDH + d0 + dR];
    }
    __syncthreads();
    for (int i = tid; i < 64 * 64; i += 256) {
        int dW = i >> 6, sW = i & 63;
        vT[((size_t)bh * CDH + d0 + dW) * CS + s0 + sW] = tile[sW][dW];
    }
}

// ---------------------------------------------------------------------------
// Attention: one wave per (b, h, 16-row q tile). Two-pass softmax with a
// padded 16x524 f32 score buffer (12-bank row stride: conflict-free scans);
// all WMMA fragments are contiguous vector loads.
// ---------------------------------------------------------------------------
#define SCP 524    // padded score-row stride (floats)

__global__ __launch_bounds__(32) void attn_kernel(
    const _Float16* __restrict__ qkv, const _Float16* __restrict__ vT,
    _Float16* __restrict__ ctx)
{
    __shared__ __align__(128) float sc[16 * SCP];
    const int lane = threadIdx.x;
    const int qt = blockIdx.x;
    const int h  = blockIdx.y;
    const int b  = blockIdx.z;
    const int lo   = lane & 15;
    const int hi16 = lane & 16;
    const int hi8  = hi16 >> 1;

    // Q fragments: 16 rows x 128 dims (4 chunks of 16x32) — 16B loads
    F16x16 qf[4];
    const _Float16* qrow = qkv + ((size_t)b * CS + qt * 16 + lo) * (3 * CE) + h * CDH;
#pragma unroll
    for (int c = 0; c < 4; ++c) {
        qf[c].h[0] = *(const v8h*)(qrow + c * 32 + hi16);
        qf[c].h[1] = *(const v8h*)(qrow + c * 32 + hi16 + 8);
    }

    // pass 1: scores (16 x 512) into LDS
    for (int kt = 0; kt < 32; ++kt) {
        v8f s;
#pragma unroll
        for (int r = 0; r < 8; ++r) s[r] = 0.0f;
        const _Float16* krow =
            qkv + ((size_t)b * CS + kt * 16 + lo) * (3 * CE) + CE + h * CDH;
#pragma unroll
        for (int c = 0; c < 4; ++c) {
            v16h kf = *(const v16h*)(krow + c * 32 + hi16);   // 32B load
            s = __builtin_amdgcn_wmma_f32_16x16x32_f16(
                false, qf[c].v, false, kf, (short)0, s, false, false);
        }
#pragma unroll
        for (int r = 0; r < 8; ++r)
            sc[(r + hi8) * SCP + kt * 16 + lo] = s[r] * 0.08838834764831845f;
    }
    __syncthreads();

    // softmax: lane -> (row = lane%16, half = lane/16 covers 256 cols)
    {
        float* rp = sc + lo * SCP + (hi16 ? 256 : 0);
        const v4f* rp4 = (const v4f*)rp;
        float mx = -3.4e38f;
        for (int i = 0; i < 64; ++i) {
            v4f t = rp4[i];
            mx = fmaxf(mx, fmaxf(fmaxf(t[0], t[1]), fmaxf(t[2], t[3])));
        }
        mx = fmaxf(mx, __shfl_xor(mx, 16));
        float sum = 0.0f;
        for (int i = 0; i < 64; ++i) {
            v4f t = rp4[i];
            v4f e;
#pragma unroll
            for (int j = 0; j < 4; ++j) { e[j] = __expf(t[j] - mx); sum += e[j]; }
            *(v4f*)(rp + i * 4) = e;
        }
        sum += __shfl_xor(sum, 16);
        float inv = 1.0f / sum;
        for (int i = 0; i < 64; ++i) {
            v4f t = rp4[i];
#pragma unroll
            for (int j = 0; j < 4; ++j) t[j] *= inv;
            *(v4f*)(rp + i * 4) = t;
        }
    }
    __syncthreads();

    // pass 2: out(16x128) = P(16x512) x V(512x128), V from vT (s contiguous)
    v8f o[8];
    for (int nt = 0; nt < 8; ++nt)
        for (int r = 0; r < 8; ++r) o[nt][r] = 0.0f;
    const _Float16* vbase = vT + (size_t)(b * CH + h) * CDH * CS;
    for (int kt = 0; kt < 16; ++kt) {
        // P fragment: two contiguous v8f LDS loads + cvt
        const float* prow = sc + lo * SCP + kt * 32;
        v8f f0 = *(const v8f*)(prow + hi16);
        v8f f1 = *(const v8f*)(prow + hi16 + 8);
        F16x16 pf;
#pragma unroll
        for (int e = 0; e < 8; ++e) {
            pf.v[e]     = (_Float16)f0[e];
            pf.v[e + 8] = (_Float16)f1[e];
        }
#pragma unroll
        for (int nt = 0; nt < 8; ++nt) {
            v16h vf = *(const v16h*)(vbase + (size_t)(nt * 16 + lo) * CS +
                                     kt * 32 + hi16);          // 32B load
            o[nt] = __builtin_amdgcn_wmma_f32_16x16x32_f16(
                false, pf.v, false, vf, (short)0, o[nt], false, false);
        }
    }
#pragma unroll
    for (int nt = 0; nt < 8; ++nt)
        for (int r = 0; r < 8; ++r) {
            size_t ci = ((size_t)b * CS + qt * 16 + r + hi8) * CE +
                        h * CDH + nt * 16 + lo;
            ctx[ci] = (_Float16)o[nt][r];
        }
}

// ---------------------------------------------------------------------------
// seg_probs: one wave per row, sigmoid(x . Wseg + bseg)
// ---------------------------------------------------------------------------
__global__ __launch_bounds__(256) void seg_kernel(
    const float* __restrict__ x, const float* __restrict__ Wseg,
    const float* __restrict__ bseg, float* __restrict__ segp)
{
    int gw = (blockIdx.x * blockDim.x + threadIdx.x) >> 5;
    int lane = threadIdx.x & 31;
    if (gw >= CB * CS) return;
    const float* row = x + (size_t)gw * CE;
    float s = 0.0f;
    for (int i = lane; i < CE; i += 32) s += row[i] * Wseg[i];
    for (int off = 16; off; off >>= 1) s += __shfl_xor(s, off);
    if (lane == 0) segp[gw] = 1.0f / (1.0f + __expf(-(s + bseg[0])));
}

// ---------------------------------------------------------------------------
// pooling: per (b, e) max / mean / var(ddof=1) over S
// ---------------------------------------------------------------------------
__global__ __launch_bounds__(256) void pool_kernel(
    const float* __restrict__ x, float* __restrict__ zmax,
    float* __restrict__ zmean, float* __restrict__ zvar)
{
    int idx = blockIdx.x * blockDim.x + threadIdx.x;
    if (idx >= CB * CE) return;
    int b = idx >> 9, e = idx & (CE - 1);
    const float* base = x + (size_t)b * CS * CE + e;
    float mx = -3.4e38f, sum = 0.0f, sq = 0.0f;
    for (int s = 0; s < CS; ++s) {
        float v = base[(size_t)s * CE];
        mx = fmaxf(mx, v);
        sum += v;
        sq  += v * v;
    }
    float mean = sum * (1.0f / CS);
    zmax[idx]  = mx;
    zmean[idx] = mean;
    zvar[idx]  = (sq - (float)CS * mean * mean) * (1.0f / (CS - 1));
}

// ---------------------------------------------------------------------------
// heads: tau = sigmoid(relu(zcat@Wt1+bt1)@Wt2+bt2),
//        cnt = softplus(relu(zmean@Wc1+bc1)@Wc2+bc2). One block per batch.
// ---------------------------------------------------------------------------
__global__ __launch_bounds__(256) void head_kernel(
    const float* __restrict__ zmax, const float* __restrict__ zmean,
    const float* __restrict__ zvar,
    const float* __restrict__ Wt1, const float* __restrict__ bt1,
    const float* __restrict__ Wt2, const float* __restrict__ bt2,
    const float* __restrict__ Wc1, const float* __restrict__ bc1,
    const float* __restrict__ Wc2, const float* __restrict__ bc2,
    float* __restrict__ outCnt, float* __restrict__ outTau)
{
    __shared__ float red[256];
    const int b = blockIdx.x, j = threadIdx.x;
    const float* zx = zmax  + (size_t)b * CE;
    const float* zm = zmean + (size_t)b * CE;
    const float* zv = zvar  + (size_t)b * CE;

    float h1 = bt1[j];
    for (int i = 0; i < CE; ++i) h1 += zx[i] * Wt1[(size_t)i * CE2 + j];
    for (int i = 0; i < CE; ++i) h1 += zm[i] * Wt1[(size_t)(CE + i) * CE2 + j];
    for (int i = 0; i < CE; ++i) h1 += zv[i] * Wt1[(size_t)(2 * CE + i) * CE2 + j];
    red[j] = fmaxf(h1, 0.0f) * Wt2[j];
    __syncthreads();
    for (int s = 128; s; s >>= 1) {
        if (j < s) red[j] += red[j + s];
        __syncthreads();
    }
    if (j == 0) outTau[b] = 1.0f / (1.0f + __expf(-(red[0] + bt2[0])));
    __syncthreads();

    float h2 = bc1[j];
    for (int i = 0; i < CE; ++i) h2 += zm[i] * Wc1[(size_t)i * CE2 + j];
    red[j] = fmaxf(h2, 0.0f) * Wc2[j];
    __syncthreads();
    for (int s = 128; s; s >>= 1) {
        if (j < s) red[j] += red[j + s];
        __syncthreads();
    }
    if (j == 0) {
        float xx = red[0] + bc2[0];
        outCnt[b] = fmaxf(xx, 0.0f) + log1pf(expf(-fabsf(xx)));  // softplus
    }
}

// ---------------------------------------------------------------------------
// DP marginals: one block per batch element. Forward/backward 6-state scans
// are serial (threads 0 and 1), marginals parallel over S.
// ---------------------------------------------------------------------------
__device__ __forceinline__ float lae(float x, float y) {
    float m = fmaxf(x, y);
    return m + log1pf(expf(fminf(x, y) - m));
}

__global__ __launch_bounds__(512) void dp_kernel(
    const float* __restrict__ segp, const int* __restrict__ wl,
    float* __restrict__ marg)
{
    __shared__ float lp[CS], lnp[CS];
    __shared__ float alph[CS + 1][CK + 1];
    __shared__ float bet[CS + 1][CK + 1];
    const int b = blockIdx.x, t = threadIdx.x;
    const int L = wl[b];
    const bool valid = t < L;
    {
        float p = segp[(size_t)b * CS + t];
        lp[t]  = valid ? logf(p + CEPS) : CNEG;
        lnp[t] = valid ? logf(1.0f - p + CEPS) : 0.0f;
    }
    __syncthreads();

    if (t == 0) {                       // forward
        float a[CK + 1];
        a[0] = 0.0f;
        for (int j = 1; j <= CK; ++j) a[j] = CNEG;
        for (int j = 0; j <= CK; ++j) alph[0][j] = a[j];
        for (int i = 1; i <= CS; ++i) {
            float lpi = lp[i - 1], lni = lnp[i - 1];
            float na[CK + 1];
            for (int j = 0; j <= CK; ++j) {
                float x1 = a[j] + lni;
                float x2 = (j > 0 ? a[j - 1] : CNEG) + lpi;
                na[j] = (j < i) ? lae(x1, x2) : CNEG;
            }
            for (int j = 0; j <= CK; ++j) { a[j] = na[j]; alph[i][j] = na[j]; }
        }
    } else if (t == 1) {                // backward
        float bb[CK + 1];
        for (int j = 0; j < CK; ++j) bb[j] = CNEG;
        bb[CK] = 0.0f;
        for (int j = 0; j <= CK; ++j) bet[CS][j] = bb[j];
        for (int i = CS - 1; i >= 0; --i) {
            float lpi = lp[i], lni = lnp[i];
            float nb[CK + 1];
            for (int j = 0; j <= CK; ++j) {
                float x1 = bb[j] + lni;
                float x2 = (j < CK ? bb[j + 1] : CNEG) + lpi;
                nb[j] = (j <= i) ? lae(x1, x2) : CNEG;
            }
            for (int j = 0; j <= CK; ++j) { bb[j] = nb[j]; bet[i][j] = nb[j]; }
        }
    }
    __syncthreads();

    const float Z = alph[CS][CK];
    float tv[CK], mx = CNEG;
    for (int j = 0; j < CK; ++j) {
        tv[j] = alph[t][j] + bet[t + 1][j + 1] + lp[t];
        mx = fmaxf(mx, tv[j]);
    }
    float s = 0.0f;
    for (int j = 0; j < CK; ++j) s += expf(tv[j] - mx);
    float lse = mx + logf(s);
    marg[(size_t)b * CS + t] = valid ? expf(lse - Z) : 0.0f;
}

// ---------------------------------------------------------------------------
extern "C" void kernel_launch(void* const* d_in, const int* in_sizes, int n_in,
                              void* d_out, int out_size, void* d_ws, size_t ws_size,
                              hipStream_t stream)
{
    (void)in_sizes; (void)n_in; (void)out_size; (void)ws_size;
    const float* enc  = (const float*)d_in[0];
    const int*   wl   = (const int*)  d_in[1];
    const float* Wqkv = (const float*)d_in[2];
    const float* bqkv = (const float*)d_in[3];
    const float* Wo   = (const float*)d_in[4];
    const float* bo   = (const float*)d_in[5];
    const float* Wseg = (const float*)d_in[6];
    const float* bseg = (const float*)d_in[7];
    const float* Wt1  = (const float*)d_in[8];
    const float* bt1  = (const float*)d_in[9];
    const float* Wt2  = (const float*)d_in[10];
    const float* bt2  = (const float*)d_in[11];
    const float* Wc1  = (const float*)d_in[12];
    const float* bc1  = (const float*)d_in[13];
    const float* Wc2  = (const float*)d_in[14];
    const float* bc2  = (const float*)d_in[15];

    // workspace layout
    char* ws = (char*)d_ws;
    const size_t M = (size_t)CB * CS;                 // 32768 rows
    _Float16* qkv_h = (_Float16*)ws;                  // M * 1536 f16
    size_t off = M * 3 * CE * sizeof(_Float16);
    _Float16* ctx_h = (_Float16*)(ws + off);          // M * 512 f16
    off += M * CE * sizeof(_Float16);
    float* xbuf = (float*)(ws + off);                 // M * 512 f32
    off += M * CE * sizeof(float);
    _Float16* vT = (_Float16*)(ws + off);             // B*H*DH*S f16
    off += (size_t)CB * CH * CDH * CS * sizeof(_Float16);
    float* zmax  = (float*)(ws + off);                // B*E each
    float* zmean = zmax  + CB * CE;
    float* zvar  = zmean + CB * CE;

    // output layout: [marg | count | tau | seg_probs]
    float* out   = (float*)d_out;
    float* marg  = out;
    float* cnt   = out + CB * CS;
    float* tau   = cnt + CB;
    float* segp  = tau + CB;

    // 1) qkv = enc @ Wqkv + bqkv  (f32 A, f16 out, no residual)
    gemm_wmma_kernel<0, 1, 0><<<dim3((3 * CE) / 64, M / 64), 128, 0, stream>>>(
        enc, Wqkv, bqkv, nullptr, qkv_h, (int)M, 3 * CE, CE);

    // 2) transpose V head slices -> vT[b][h][d][s]
    vtrans_kernel<<<dim3(CS / 64, CDH / 64, CB * CH), 256, 0, stream>>>(qkv_h, vT);

    // 3) multi-head attention -> ctx (f16)
    attn_kernel<<<dim3(CS / 16, CH, CB), 32, 0, stream>>>(qkv_h, vT, ctx_h);

    // 4) x = enc + ctx @ Wo + bo  (f16 A, f32 out, residual)
    gemm_wmma_kernel<1, 0, 1><<<dim3(CE / 64, M / 64), 128, 0, stream>>>(
        ctx_h, Wo, bo, enc, xbuf, (int)M, CE, CE);

    // 5) seg_probs
    seg_kernel<<<(int)(M / 8), 256, 0, stream>>>(xbuf, Wseg, bseg, segp);

    // 6) pooling
    pool_kernel<<<(CB * CE) / 256, 256, 0, stream>>>(xbuf, zmax, zmean, zvar);

    // 7) heads
    head_kernel<<<CB, 256, 0, stream>>>(zmax, zmean, zvar,
                                        Wt1, bt1, Wt2, bt2,
                                        Wc1, bc1, Wc2, bc2, cnt, tau);

    // 8) DP marginals
    dp_kernel<<<CB, CS, 0, stream>>>(segp, wl, marg);
}
